// CrossAttention_59107339928124
// MI455X (gfx1250) — matrix operand order repaired and checked by reading, not compiled, not run on hardware
//
#include <hip/hip_runtime.h>
#include <hip/hip_bf16.h>

typedef __bf16 bf16;
typedef __attribute__((ext_vector_type(16))) __bf16 v16bf;
typedef __attribute__((ext_vector_type(8)))  __bf16 v8bf;
typedef __attribute__((ext_vector_type(4)))  __bf16 v4bf;
typedef __attribute__((ext_vector_type(8)))  float  v8f;
typedef __attribute__((ext_vector_type(4)))  float  v4f;

#define DIM   1024
#define SEQ   2048
#define BATCH 16
#define LDSW  40   // padded LDS row stride in bf16 elems (80B, 16B aligned)

// Block tile: 128(M) x 128(N) x 32(K). 8 waves in 4(M) x 2(N); wave tile 32x64.
// Each K-step per wave: 2 A-frags + 4 B-frags -> 8 v_wmma_f32_16x16x32_bf16.

// ---------------- WMMA helpers (CDNA5, wave32) ----------------

__device__ inline v8f vzero8() {
    v8f z;
#pragma unroll
    for (int i = 0; i < 8; ++i) z[i] = 0.0f;
    return z;
}

__device__ inline v8f wmma_bf16(v16bf a, v16bf b, v8f c) {
    // (neg_a, A, neg_b, B, c_mod, C, reuse_a, reuse_b)
    return __builtin_amdgcn_wmma_f32_16x16x32_bf16(false, a, false, b, (short)0, c, false, false);
}

// A fragment: 16x32 bf16, LDS row-major [M][K], stride LDSW.
// Lane L<16: row=L, K = 0..7 & 16..23 ; lane L>=16: row=L-16, K = 8..15 & 24..31.
__device__ inline v16bf load_frag_a(const bf16* As, int lane) {
    const int row = lane & 15;
    const int kh  = (lane >> 4) << 3;
    const v8bf lo = *(const v8bf*)(As + row * LDSW + kh);
    const v8bf hi = *(const v8bf*)(As + row * LDSW + kh + 16);
    v16bf a;
#pragma unroll
    for (int i = 0; i < 8; ++i) { a[i] = lo[i]; a[i + 8] = hi[i]; }
    return a;
}

// B fragment: 32x16 bf16, LDS stored K-major [N][K], stride LDSW.
// Lane L<16: col=L, K = 0..7 & 16..23 ; lane L>=16: col=L-16, K = 8..15 & 24..31.
__device__ inline v16bf load_frag_b(const bf16* Bs, int lane) {
    const int col = lane & 15;
    const int kb  = (lane >> 4) << 3;
    const v8bf lo = *(const v8bf*)(Bs + col * LDSW + kb);
    const v8bf hi = *(const v8bf*)(Bs + col * LDSW + kb + 16);
    v16bf b;
#pragma unroll
    for (int i = 0; i < 8; ++i) { b[i] = lo[i]; b[i + 8] = hi[i]; }
    return b;
}

// One K-step for a wave: 8 back-to-back WMMAs on its 32x64 tile.
__device__ inline void mma_step(const bf16* As, const bf16* Bs, int wm, int wn,
                                int lane, v8f (&acc)[2][4]) {
    v16bf a[2], b[4];
#pragma unroll
    for (int i = 0; i < 2; ++i) a[i] = load_frag_a(As + (wm * 32 + i * 16) * LDSW, lane);
#pragma unroll
    for (int j = 0; j < 4; ++j) b[j] = load_frag_b(Bs + (wn * 64 + j * 16) * LDSW, lane);
#pragma unroll
    for (int i = 0; i < 2; ++i)
#pragma unroll
        for (int j = 0; j < 4; ++j) acc[i][j] = wmma_bf16(a[i], b[j], acc[i][j]);
}

// ---------------- Kernel 1: fused Q/K/V projection ----------------
// grid = (DIM/128, (B*SEQ)/128, 3), block = 256

__device__ inline void qkv_load(const float* A, const float* W, int m0, int n0, int k0,
                                int tid, v4bf (&aR)[4], v4bf (&bR)[4]) {
#pragma unroll
    for (int it = 0; it < 4; ++it) {
        const int idx = tid + it * 256;            // 1024 float4 chunks each
        const int row = idx >> 3, c4 = idx & 7;    // A: 128 rows x 8 chunks
        const v4f v = *(const v4f*)(A + (size_t)(m0 + row) * DIM + k0 + c4 * 4);
#pragma unroll
        for (int e = 0; e < 4; ++e) aR[it][e] = (bf16)v[e];
        const int k = idx >> 5, c = idx & 31;      // B: 32 k-rows x 32 chunks
        const v4f w = *(const v4f*)(W + (size_t)(k0 + k) * DIM + n0 + c * 4);
#pragma unroll
        for (int e = 0; e < 4; ++e) bR[it][e] = (bf16)w[e];
    }
}

__device__ inline void qkv_store(bf16* As, bf16* Bs, int tid,
                                 const v4bf (&aR)[4], const v4bf (&bR)[4]) {
#pragma unroll
    for (int it = 0; it < 4; ++it) {
        const int idx = tid + it * 256;
        const int row = idx >> 3, c4 = idx & 7;
        *(v4bf*)(As + row * LDSW + c4 * 4) = aR[it];
        const int k = idx >> 5, c = idx & 31;      // transpose W into [n][k]
#pragma unroll
        for (int e = 0; e < 4; ++e) Bs[(c * 4 + e) * LDSW + k] = bR[it][e];
    }
}

__global__ __launch_bounds__(256)
void qkv_kernel(const float* __restrict__ x,  const float* __restrict__ enc,
                const float* __restrict__ Wq, const float* __restrict__ bq,
                const float* __restrict__ Wk, const float* __restrict__ bk,
                const float* __restrict__ Wv, const float* __restrict__ bv,
                bf16* __restrict__ Qo, bf16* __restrict__ Ko, bf16* __restrict__ Vo)
{
    const int which = blockIdx.z;
    const float* A    = (which == 0) ? x  : enc;
    const float* W    = (which == 0) ? Wq : (which == 1 ? Wk : Wv);
    const float* bias = (which == 0) ? bq : (which == 1 ? bk : bv);
    bf16* out         = (which == 0) ? Qo : (which == 1 ? Ko : Vo);

    const int m0 = blockIdx.y * 128;
    const int n0 = blockIdx.x * 128;

    __shared__ bf16 As[2][128 * LDSW];
    __shared__ bf16 Bs[2][128 * LDSW];

    const int tid  = threadIdx.x;
    const int wave = tid >> 5, lane = tid & 31;
    const int wm = wave & 3, wn = wave >> 2;

    v8f acc[2][4];
#pragma unroll
    for (int i = 0; i < 2; ++i)
#pragma unroll
        for (int j = 0; j < 4; ++j) acc[i][j] = vzero8();

    v4bf aR[4], bR[4];
    qkv_load(A, W, m0, n0, 0, tid, aR, bR);
    qkv_store(As[0], Bs[0], tid, aR, bR);
    __syncthreads();

    int buf = 0;
    for (int k0 = 0; k0 < DIM; k0 += 32, buf ^= 1) {
        const bool more = (k0 + 32 < DIM);
        if (more) qkv_load(A, W, m0, n0, k0 + 32, tid, aR, bR);   // global -> regs
        mma_step(As[buf], Bs[buf], wm, wn, lane, acc);            // WMMA hides latency
        if (more) qkv_store(As[buf ^ 1], Bs[buf ^ 1], tid, aR, bR);
        __syncthreads();
    }

    // Epilogue: bias add, bf16 store. C layout: VGPR e -> M = e + 8*(lane>=16), N = lane&15
    const int colL  = lane & 15;
    const int rbase = (lane >> 4) << 3;
#pragma unroll
    for (int i2 = 0; i2 < 2; ++i2)
#pragma unroll
        for (int j2 = 0; j2 < 4; ++j2) {
            const int mB = m0 + wm * 32 + i2 * 16;
            const int nB = n0 + wn * 64 + j2 * 16;
            const float bv_ = bias[nB + colL];
#pragma unroll
            for (int e = 0; e < 8; ++e)
                out[(size_t)(mB + rbase + e) * DIM + nB + colL] = (bf16)(acc[i2][j2][e] + bv_);
        }
}

// ---------------- Kernel 2: S = Q @ K^T * scale ----------------
// grid = (SEQ/128, SEQ/128, BATCH)

__device__ inline void sc_load(const bf16* Ab, const bf16* Bb, int m0, int n0, int k0,
                               int tid, v8bf (&aR)[2], v8bf (&bR)[2]) {
#pragma unroll
    for (int it = 0; it < 2; ++it) {
        const int idx = tid + it * 256;            // 512 v8bf chunks each
        const int row = idx >> 2, c = idx & 3;
        aR[it] = *(const v8bf*)(Ab + (size_t)(m0 + row) * DIM + k0 + c * 8);
        bR[it] = *(const v8bf*)(Bb + (size_t)(n0 + row) * DIM + k0 + c * 8);
    }
}

__device__ inline void sc_store(bf16* As, bf16* Bs, int tid,
                                const v8bf (&aR)[2], const v8bf (&bR)[2]) {
#pragma unroll
    for (int it = 0; it < 2; ++it) {
        const int idx = tid + it * 256;
        const int row = idx >> 2, c = idx & 3;
        *(v8bf*)(As + row * LDSW + c * 8) = aR[it];
        *(v8bf*)(Bs + row * LDSW + c * 8) = bR[it];   // K rows are already [N][K]
    }
}

__global__ __launch_bounds__(256)
void scores_kernel(const bf16* __restrict__ Q, const bf16* __restrict__ Kc,
                   bf16* __restrict__ S)
{
    const int b  = blockIdx.z;
    const int m0 = blockIdx.y * 128;
    const int n0 = blockIdx.x * 128;

    const bf16* Ab = Q  + (size_t)b * SEQ * DIM;
    const bf16* Bb = Kc + (size_t)b * SEQ * DIM;
    bf16*       Sb = S  + (size_t)b * SEQ * SEQ;

    __shared__ bf16 As[2][128 * LDSW];
    __shared__ bf16 Bs[2][128 * LDSW];

    const int tid  = threadIdx.x;
    const int wave = tid >> 5, lane = tid & 31;
    const int wm = wave & 3, wn = wave >> 2;

    v8f acc[2][4];
#pragma unroll
    for (int i = 0; i < 2; ++i)
#pragma unroll
        for (int j = 0; j < 4; ++j) acc[i][j] = vzero8();

    v8bf aR[2], bR[2];
    sc_load(Ab, Bb, m0, n0, 0, tid, aR, bR);
    sc_store(As[0], Bs[0], tid, aR, bR);
    __syncthreads();

    int buf = 0;
    for (int k0 = 0; k0 < DIM; k0 += 32, buf ^= 1) {
        const bool more = (k0 + 32 < DIM);
        if (k0 + 64 < DIM)
            __builtin_prefetch(Ab + (size_t)(m0 + (tid >> 2)) * DIM + k0 + 64, 0, 0);
        if (more) sc_load(Ab, Bb, m0, n0, k0 + 32, tid, aR, bR);
        mma_step(As[buf], Bs[buf], wm, wn, lane, acc);
        if (more) sc_store(As[buf ^ 1], Bs[buf ^ 1], tid, aR, bR);
        __syncthreads();
    }

    const float scale = 0.03125f;  // 1/sqrt(1024)
    const int colL  = lane & 15;
    const int rbase = (lane >> 4) << 3;
#pragma unroll
    for (int i2 = 0; i2 < 2; ++i2)
#pragma unroll
        for (int j2 = 0; j2 < 4; ++j2) {
            const int mB = m0 + wm * 32 + i2 * 16;
            const int nB = n0 + wn * 64 + j2 * 16;
#pragma unroll
            for (int e = 0; e < 8; ++e)
                Sb[(size_t)(mB + rbase + e) * SEQ + nB + colL] = (bf16)(acc[i2][j2][e] * scale);
        }
}

// ---------------- Kernel 3: in-place row softmax over S ----------------
__global__ __launch_bounds__(256)
void softmax_kernel(bf16* __restrict__ S)
{
    bf16* p = S + (size_t)blockIdx.x * SEQ;
    const int tid = threadIdx.x;

    float v[8];
    float m = -1e30f;
#pragma unroll
    for (int i = 0; i < 8; ++i) {
        v[i] = (float)p[tid + i * 256];
        m = fmaxf(m, v[i]);
    }
#pragma unroll
    for (int off = 16; off > 0; off >>= 1) m = fmaxf(m, __shfl_xor(m, off, 32));

    __shared__ float redmax[8];
    __shared__ float redsum[8];
    if ((tid & 31) == 0) redmax[tid >> 5] = m;
    __syncthreads();
    float bm = redmax[0];
#pragma unroll
    for (int i = 1; i < 8; ++i) bm = fmaxf(bm, redmax[i]);

    float s = 0.0f;
#pragma unroll
    for (int i = 0; i < 8; ++i) { v[i] = __expf(v[i] - bm); s += v[i]; }
#pragma unroll
    for (int off = 16; off > 0; off >>= 1) s += __shfl_xor(s, off, 32);
    if ((tid & 31) == 0) redsum[tid >> 5] = s;
    __syncthreads();
    float tot = 0.0f;
#pragma unroll
    for (int i = 0; i < 8; ++i) tot += redsum[i];

    const float inv = 1.0f / tot;
#pragma unroll
    for (int i = 0; i < 8; ++i) p[tid + i * 256] = (bf16)(v[i] * inv);
}

// ---------------- Kernel 4: H = P @ V (f32 out) ----------------
// grid = (DIM/128, SEQ/128, BATCH)

__device__ inline void pv_load(const bf16* Ab, const bf16* Vb, int m0, int n0, int k0,
                               int tid, v8bf (&aR)[2], v8bf (&bR)[2]) {
#pragma unroll
    for (int it = 0; it < 2; ++it) {
        const int idx = tid + it * 256;
        const int row = idx >> 2, c = idx & 3;                 // A: 128 rows x 4 chunks
        aR[it] = *(const v8bf*)(Ab + (size_t)(m0 + row) * SEQ + k0 + c * 8);
        const int k = idx >> 4, cc = idx & 15;                 // V: 32 k-rows x 16 chunks
        bR[it] = *(const v8bf*)(Vb + (size_t)(k0 + k) * DIM + n0 + cc * 8);
    }
}

__device__ inline void pv_store(bf16* As, bf16* Bs, int tid,
                                const v8bf (&aR)[2], const v8bf (&bR)[2]) {
#pragma unroll
    for (int it = 0; it < 2; ++it) {
        const int idx = tid + it * 256;
        const int row = idx >> 2, c = idx & 3;
        *(v8bf*)(As + row * LDSW + c * 8) = aR[it];
        const int k = idx >> 4, cc = idx & 15;                 // transpose V into [n][k]
#pragma unroll
        for (int e = 0; e < 8; ++e) Bs[(cc * 8 + e) * LDSW + k] = bR[it][e];
    }
}

__global__ __launch_bounds__(256)
void pv_kernel(const bf16* __restrict__ P, const bf16* __restrict__ V,
               float* __restrict__ H)
{
    const int b  = blockIdx.z;
    const int m0 = blockIdx.y * 128;
    const int n0 = blockIdx.x * 128;

    const bf16* Ab = P + (size_t)b * SEQ * SEQ;
    const bf16* Vb = V + (size_t)b * SEQ * DIM;
    float*      Hb = H + (size_t)b * SEQ * DIM;

    __shared__ bf16 As[2][128 * LDSW];
    __shared__ bf16 Bs[2][128 * LDSW];

    const int tid  = threadIdx.x;
    const int wave = tid >> 5, lane = tid & 31;
    const int wm = wave & 3, wn = wave >> 2;

    v8f acc[2][4];
#pragma unroll
    for (int i = 0; i < 2; ++i)
#pragma unroll
        for (int j = 0; j < 4; ++j) acc[i][j] = vzero8();

    v8bf aR[2], bR[2];
    pv_load(Ab, Vb, m0, n0, 0, tid, aR, bR);
    pv_store(As[0], Bs[0], tid, aR, bR);
    __syncthreads();

    int buf = 0;
    for (int k0 = 0; k0 < SEQ; k0 += 32, buf ^= 1) {
        const bool more = (k0 + 32 < SEQ);
        if (more) pv_load(Ab, Vb, m0, n0, k0 + 32, tid, aR, bR);
        mma_step(As[buf], Bs[buf], wm, wn, lane, acc);
        if (more) pv_store(As[buf ^ 1], Bs[buf ^ 1], tid, aR, bR);
        __syncthreads();
    }

    const int colL  = lane & 15;
    const int rbase = (lane >> 4) << 3;
#pragma unroll
    for (int i2 = 0; i2 < 2; ++i2)
#pragma unroll
        for (int j2 = 0; j2 < 4; ++j2) {
            const int mB = m0 + wm * 32 + i2 * 16;
            const int nB = n0 + wn * 64 + j2 * 16;
#pragma unroll
            for (int e = 0; e < 8; ++e)
                Hb[(size_t)(mB + rbase + e) * DIM + nB + colL] = acc[i2][j2][e];
        }
}

// ---------------- Host launch ----------------

extern "C" void kernel_launch(void* const* d_in, const int* in_sizes, int n_in,
                              void* d_out, int out_size, void* d_ws, size_t ws_size,
                              hipStream_t stream) {
    const float* x   = (const float*)d_in[0];
    const float* enc = (const float*)d_in[1];
    const float* Wq  = (const float*)d_in[2];
    const float* bq  = (const float*)d_in[3];
    const float* Wk  = (const float*)d_in[4];
    const float* bk  = (const float*)d_in[5];
    const float* Wv  = (const float*)d_in[6];
    const float* bv  = (const float*)d_in[7];

    const size_t QKV_ELEMS = (size_t)BATCH * SEQ * DIM;
    bf16* Q = (bf16*)d_ws;
    bf16* K = Q + QKV_ELEMS;
    bf16* V = K + QKV_ELEMS;
    bf16* S = V + QKV_ELEMS;      // BATCH*SEQ*SEQ bf16
    float* H = (float*)d_out;

    qkv_kernel<<<dim3(DIM / 128, (BATCH * SEQ) / 128, 3), 256, 0, stream>>>(
        x, enc, Wq, bq, Wk, bk, Wv, bv, Q, K, V);
    scores_kernel<<<dim3(SEQ / 128, SEQ / 128, BATCH), 256, 0, stream>>>(Q, K, S);
    softmax_kernel<<<dim3(BATCH * SEQ), 256, 0, stream>>>(S);
    pv_kernel<<<dim3(DIM / 128, SEQ / 128, BATCH), 256, 0, stream>>>(S, V, H);
}
